// FastAttention_31714038513717
// MI455X (gfx1250) — compile-verified, matrix operand order checked by reading, not run on hardware
//
#include <hip/hip_runtime.h>
#include <hip/hip_bf16.h>

#define DEV __device__ __forceinline__

typedef __bf16 bf16;
typedef __attribute__((ext_vector_type(16))) __bf16 v16bf;
typedef __attribute__((ext_vector_type(8)))  __bf16 v8bf;
typedef __attribute__((ext_vector_type(8)))  float  v8f;
typedef __attribute__((ext_vector_type(4)))  float  v4f;

// Problem constants
#define BATCH 8
#define SEQ   4096
#define CH    512
#define HEADS 8
#define HDIM  64
#define ROWS  (BATCH * SEQ)        // 32768
#define QKVC  (3 * CH)             // 1536
#define SCALE_HD 0.125f            // 64^-0.5

// ---------------------------------------------------------------------------
// Fragment loaders.  CDNA5 16-bit A-matrix 16x32 layout (ISA 7.12.2):
//   lane<16 : row M=lane,    elems 0..7 <-> K=0..7,   elems 8..15 <-> K=16..23
//   lane>=16: row M=lane-16, elems 0..7 <-> K=8..15,  elems 8..15 <-> K=24..31
// B (32x16, N in lanes) mirrors the same K pattern, so both x (row-major
// [M,K]) and w (row-major [N,K]) load with the same contiguous-K gather.
// ---------------------------------------------------------------------------

DEV v16bf frag_from_f32(const float* __restrict__ row_base, int ld, int lane) {
  const int r = lane & 15, half = lane >> 4;
  const float* p = row_base + (size_t)r * ld + half * 8;
  v4f a = *(const v4f*)(p);
  v4f b = *(const v4f*)(p + 4);
  v4f c = *(const v4f*)(p + 16);
  v4f d = *(const v4f*)(p + 20);
  v16bf o;
#pragma unroll
  for (int i = 0; i < 4; ++i) {
    o[i]      = (bf16)a[i];
    o[4 + i]  = (bf16)b[i];
    o[8 + i]  = (bf16)c[i];
    o[12 + i] = (bf16)d[i];
  }
  return o;
}

DEV v16bf frag_from_bf16(const bf16* __restrict__ row_base, int ld, int lane) {
  const int r = lane & 15, half = lane >> 4;
  const bf16* p = row_base + (size_t)r * ld + half * 8;
  v8bf a = *(const v8bf*)(p);
  v8bf b = *(const v8bf*)(p + 16);
  v16bf o;
#pragma unroll
  for (int i = 0; i < 8; ++i) { o[i] = a[i]; o[8 + i] = b[i]; }
  return o;
}

// A = v(bf16) scaled per-channel by f32 vector (u = global_k * v)
DEV v16bf frag_from_bf16_scaled(const bf16* __restrict__ row_base, int ld,
                                const float* __restrict__ scale_base, int lane) {
  const int r = lane & 15, half = lane >> 4;
  const bf16* p = row_base + (size_t)r * ld + half * 8;
  const float* s = scale_base + half * 8;
  v8bf a = *(const v8bf*)(p);
  v8bf b = *(const v8bf*)(p + 16);
  v4f s0 = *(const v4f*)(s);
  v4f s1 = *(const v4f*)(s + 4);
  v4f s2 = *(const v4f*)(s + 16);
  v4f s3 = *(const v4f*)(s + 20);
  v16bf o;
#pragma unroll
  for (int i = 0; i < 4; ++i) {
    o[i]      = (bf16)((float)a[i]     * s0[i]);
    o[4 + i]  = (bf16)((float)a[4 + i] * s1[i]);
    o[8 + i]  = (bf16)((float)b[i]     * s2[i]);
    o[12 + i] = (bf16)((float)b[4 + i] * s3[i]);
  }
  return o;
}

DEV v8f wmma_bf16(v16bf a, v16bf b, v8f c) {
  return __builtin_amdgcn_wmma_f32_16x16x32_bf16(false, a, false, b,
                                                 (short)0, c, false, false);
}

// ---------------------------------------------------------------------------
// f32 -> bf16 conversion (weights)
// ---------------------------------------------------------------------------
__global__ __launch_bounds__(256) void cvt_bf16_kernel(const float* __restrict__ in,
                                                       bf16* __restrict__ out, int n) {
  int i = blockIdx.x * 256 + threadIdx.x;
  if (i < n) out[i] = (bf16)in[i];
}

// ---------------------------------------------------------------------------
// GEMM1: qkv_bf[bn, d] = sum_c x[bn, c] * w_qkv[d, c]   (bf16 out, f32 accum)
// Per wave: 32 rows x 64 cols, register double-buffered K pipeline.
// Grid: (QKVC/256 = 6, ROWS/64 = 512), 8 waves/block.
// ---------------------------------------------------------------------------
__global__ __launch_bounds__(256) void gemm_qkv_kernel(const float* __restrict__ x,
                                                       const bf16* __restrict__ wq,
                                                       bf16* __restrict__ qkv) {
  const int lane = threadIdx.x & 31;
  const int wave = threadIdx.x >> 5;
  const int m0 = blockIdx.y * 64 + (wave >> 2) * 32;
  const int n0 = blockIdx.x * 256 + (wave & 3) * 64;

  v8f acc[2][4] = {};
  const float* xbase = x  + (size_t)m0 * CH;
  const bf16*  wbase = wq + (size_t)n0 * CH;

  // prologue: fragments for k = 0
  v16bf a0 = frag_from_f32(xbase,                   CH, lane);
  v16bf a1 = frag_from_f32(xbase + (size_t)16 * CH, CH, lane);
  v16bf bb[4];
#pragma unroll
  for (int j = 0; j < 4; ++j)
    bb[j] = frag_from_bf16(wbase + (size_t)(j * 16) * CH, CH, lane);

#pragma unroll 4
  for (int k = 0; k < CH; k += 32) {
    const int kn = (k + 32 < CH) ? (k + 32) : 0;   // wrap on last iter (discarded)
    __builtin_prefetch(xbase + ((size_t)(lane & 15)) * CH + kn + 32, 0, 3);

    // issue next iteration's loads before consuming current fragments
    v16bf na0 = frag_from_f32(xbase + kn,                   CH, lane);
    v16bf na1 = frag_from_f32(xbase + (size_t)16 * CH + kn, CH, lane);
    v16bf nbb[4];
#pragma unroll
    for (int j = 0; j < 4; ++j)
      nbb[j] = frag_from_bf16(wbase + (size_t)(j * 16) * CH + kn, CH, lane);

#pragma unroll
    for (int j = 0; j < 4; ++j) {
      acc[0][j] = wmma_bf16(a0, bb[j], acc[0][j]);
      acc[1][j] = wmma_bf16(a1, bb[j], acc[1][j]);
    }

    a0 = na0; a1 = na1;
#pragma unroll
    for (int j = 0; j < 4; ++j) bb[j] = nbb[j];
  }

  // C/D layout: lane L, VGPR v -> row = v + (L>>4)*8, col = L&15
  const int r15 = lane & 15, half = lane >> 4;
#pragma unroll
  for (int i = 0; i < 2; ++i) {
#pragma unroll
    for (int j = 0; j < 4; ++j) {
      const int col = n0 + j * 16 + r15;
#pragma unroll
      for (int v = 0; v < 8; ++v) {
        const int row = m0 + i * 16 + half * 8 + v;
        qkv[(size_t)row * QKVC + col] = (bf16)acc[i][j][v];
      }
    }
  }
}

// ---------------------------------------------------------------------------
// Softmax attention pooling.  One block per (b,h).
//   logits[n] = SCALE * sum_c data[n,c] * (wh[h,c] * premul[b,h,c]?)
//   outg[b,h,c] = sum_n softmax(logits)[n] * data[n,c]
// data = qkv_bf at channel offset chanOff (0 -> q, 512 -> k)
// ---------------------------------------------------------------------------
__global__ __launch_bounds__(256) void pool_kernel(const bf16* __restrict__ qkv,
                                                   int chanOff,
                                                   const float* __restrict__ wh,
                                                   const float* __restrict__ premul,
                                                   float* __restrict__ outg) {
  __shared__ float s_logit[SEQ];
  __shared__ float s_red[256];
  __shared__ float s_w[HDIM];

  const int t = threadIdx.x;
  const int bh = blockIdx.x;
  const int b = bh >> 3, h = bh & 7;

  if (t < HDIM) {
    float w = wh[h * HDIM + t];
    if (premul) w *= premul[b * CH + h * HDIM + t];
    s_w[t] = w;
  }
  __syncthreads();

  const bf16* base = qkv + (size_t)b * SEQ * QKVC + chanOff + h * HDIM;

  // pass 1: logits + local max
  float lmax = -3.0e38f;
  for (int n = t; n < SEQ; n += 256) {
    const bf16* p = base + (size_t)n * QKVC;
    float dot = 0.f;
#pragma unroll
    for (int c8 = 0; c8 < HDIM; c8 += 8) {
      v8bf vv = *(const v8bf*)(p + c8);
#pragma unroll
      for (int i = 0; i < 8; ++i) dot += (float)vv[i] * s_w[c8 + i];
    }
    dot *= SCALE_HD;
    s_logit[n] = dot;
    lmax = fmaxf(lmax, dot);
  }
  s_red[t] = lmax;
  __syncthreads();
  for (int s = 128; s > 0; s >>= 1) {
    if (t < s) s_red[t] = fmaxf(s_red[t], s_red[t + s]);
    __syncthreads();
  }
  const float gmax = s_red[0];
  __syncthreads();

  // pass 2: exp + local sum
  float lsum = 0.f;
  for (int n = t; n < SEQ; n += 256) {
    float e = __expf(s_logit[n] - gmax);
    s_logit[n] = e;
    lsum += e;
  }
  s_red[t] = lsum;
  __syncthreads();
  for (int s = 128; s > 0; s >>= 1) {
    if (t < s) s_red[t] += s_red[t + s];
    __syncthreads();
  }
  const float inv = 1.0f / s_red[0];
  __syncthreads();

  // pass 3: weighted sum over n, 4 partials per channel
  const int hd = t & 63, g = t >> 6;
  float accv = 0.f;
  for (int n = g; n < SEQ; n += 4)
    accv += s_logit[n] * (float)base[(size_t)n * QKVC + hd];
  s_red[t] = accv;
  __syncthreads();
  if (t < HDIM)
    outg[b * CH + h * HDIM + t] =
        (s_red[t] + s_red[t + 64] + s_red[t + 128] + s_red[t + 192]) * inv;
}

// ---------------------------------------------------------------------------
// GEMM2: out[bn, d] = sum_c (gk[b,c]*v[bn,c]) * w_proj[d, c] + b_proj[d] + q[bn, d]
// Grid: (CH/256 = 2, ROWS/64 = 512), 8 waves/block, 32x64 tile per wave,
// register double-buffered K pipeline.
// ---------------------------------------------------------------------------
__global__ __launch_bounds__(256) void gemm_out_kernel(const bf16* __restrict__ qkv,
                                                       const bf16* __restrict__ wp,
                                                       const float* __restrict__ bproj,
                                                       const float* __restrict__ gk,
                                                       float* __restrict__ out) {
  const int lane = threadIdx.x & 31;
  const int wave = threadIdx.x >> 5;
  const int m0 = blockIdx.y * 64 + (wave >> 2) * 32;
  const int n0 = blockIdx.x * 256 + (wave & 3) * 64;
  const int b = m0 >> 12;                      // m0 / 4096

  const float* gkb = gk + b * CH;
  v8f acc[2][4] = {};
  const bf16* vbase = qkv + (size_t)m0 * QKVC + 2 * CH;  // v plane
  const bf16* wbase = wp + (size_t)n0 * CH;

  v16bf a0 = frag_from_bf16_scaled(vbase,                     QKVC, gkb, lane);
  v16bf a1 = frag_from_bf16_scaled(vbase + (size_t)16 * QKVC, QKVC, gkb, lane);
  v16bf bb[4];
#pragma unroll
  for (int j = 0; j < 4; ++j)
    bb[j] = frag_from_bf16(wbase + (size_t)(j * 16) * CH, CH, lane);

#pragma unroll 4
  for (int k = 0; k < CH; k += 32) {
    const int kn = (k + 32 < CH) ? (k + 32) : 0;
    __builtin_prefetch(vbase + ((size_t)(lane & 15)) * QKVC + kn + 32, 0, 3);

    v16bf na0 = frag_from_bf16_scaled(vbase + kn,                     QKVC, gkb + kn, lane);
    v16bf na1 = frag_from_bf16_scaled(vbase + (size_t)16 * QKVC + kn, QKVC, gkb + kn, lane);
    v16bf nbb[4];
#pragma unroll
    for (int j = 0; j < 4; ++j)
      nbb[j] = frag_from_bf16(wbase + (size_t)(j * 16) * CH + kn, CH, lane);

#pragma unroll
    for (int j = 0; j < 4; ++j) {
      acc[0][j] = wmma_bf16(a0, bb[j], acc[0][j]);
      acc[1][j] = wmma_bf16(a1, bb[j], acc[1][j]);
    }

    a0 = na0; a1 = na1;
#pragma unroll
    for (int j = 0; j < 4; ++j) bb[j] = nbb[j];
  }

  // epilogue: + b_proj + q   (q is channel [0,512) of qkv)
  const int r15 = lane & 15, half = lane >> 4;
#pragma unroll
  for (int i = 0; i < 2; ++i) {
#pragma unroll
    for (int j = 0; j < 4; ++j) {
      const int col = n0 + j * 16 + r15;
      const float bias = bproj[col];
#pragma unroll
      for (int v = 0; v < 8; ++v) {
        const int row = m0 + i * 16 + half * 8 + v;
        const float qv = (float)qkv[(size_t)row * QKVC + col];
        out[(size_t)row * CH + col] = acc[i][j][v] + bias + qv;
      }
    }
  }
}

// ---------------------------------------------------------------------------
// Launch
// inputs: 0=x [8,4096,512] f32, 1=w_qkv [1536,512] f32, 2=w_proj [512,512] f32,
//         3=b_proj [512] f32, 4=w_q [8,64] f32, 5=w_k [8,64] f32
// output: [8,4096,512] f32
// ---------------------------------------------------------------------------
extern "C" void kernel_launch(void* const* d_in, const int* in_sizes, int n_in,
                              void* d_out, int out_size, void* d_ws, size_t ws_size,
                              hipStream_t stream) {
  const float* x      = (const float*)d_in[0];
  const float* w_qkv  = (const float*)d_in[1];
  const float* w_proj = (const float*)d_in[2];
  const float* b_proj = (const float*)d_in[3];
  const float* w_q    = (const float*)d_in[4];
  const float* w_k    = (const float*)d_in[5];
  float* out = (float*)d_out;

  char* ws = (char*)d_ws;
  size_t o = 0;
  bf16* qkv_bf   = (bf16*)(ws + o); o += (size_t)ROWS * QKVC * sizeof(bf16);  // 96 MB
  bf16* wqkv_bf  = (bf16*)(ws + o); o += (size_t)QKVC * CH * sizeof(bf16);
  bf16* wproj_bf = (bf16*)(ws + o); o += (size_t)CH * CH * sizeof(bf16);
  float* gq      = (float*)(ws + o); o += (size_t)BATCH * CH * sizeof(float);
  float* gk      = (float*)(ws + o); o += (size_t)BATCH * CH * sizeof(float);
  (void)o; (void)ws_size; (void)in_sizes; (void)n_in; (void)out_size;

  const int nwqkv  = QKVC * CH;   // 786432
  const int nwproj = CH * CH;     // 262144
  cvt_bf16_kernel<<<(nwqkv + 255) / 256, 256, 0, stream>>>(w_qkv, wqkv_bf, nwqkv);
  cvt_bf16_kernel<<<(nwproj + 255) / 256, 256, 0, stream>>>(w_proj, wproj_bf, nwproj);

  gemm_qkv_kernel<<<dim3(QKVC / 256, ROWS / 64), 256, 0, stream>>>(x, wqkv_bf, qkv_bf);

  pool_kernel<<<BATCH * HEADS, 256, 0, stream>>>(qkv_bf, 0,  w_q, nullptr, gq);
  pool_kernel<<<BATCH * HEADS, 256, 0, stream>>>(qkv_bf, CH, w_k, gq,      gk);

  gemm_out_kernel<<<dim3(CH / 256, ROWS / 64), 256, 0, stream>>>(qkv_bf, wproj_bf,
                                                                 b_proj, gk, out);
}
